// Attention_13030930776064
// MI455X (gfx1250) — matrix operand order repaired and checked by reading, not compiled, not run on hardware
//
#include <hip/hip_runtime.h>
#include <hip/hip_bf16.h>

// Problem constants (reference): B=32, S=2048, E=A=D=1024
#define Bn 32
#define Sn 2048
#define En 1024
#define An 1024

typedef __attribute__((ext_vector_type(16))) __bf16 v16bf;
typedef __attribute__((ext_vector_type(8)))  float  v8f;

// f32 -> bf16 pair packed into one u32 (lo in [15:0]).
// The two-operand packed convert exists on gfx1250 (the compiler itself emits
// "v_cvt_pk_bf16_f32 vD, vS0, vS1"), but no clang builtin exposes it on this
// toolchain and scalar casts lower to ~5 VALU ops of merge code per pair.
// Inline asm gets the 1-instruction form (RTNE per ISA).
__device__ __forceinline__ unsigned pack_bf16(float lo, float hi) {
    unsigned r;
    asm("v_cvt_pk_bf16_f32 %0, %1, %2" : "=v"(r) : "v"(lo), "v"(hi));
    return r;
}

// ---------------------------------------------------------------------------
// K1: x2[b,a] = dinput[b,:] . W_dec[:,a] + b_dec[a]      (32 x 1024, tiny)
// ---------------------------------------------------------------------------
__global__ __launch_bounds__(256) void dec_proj_kernel(
    const float* __restrict__ dinput, const float* __restrict__ W_dec,
    const float* __restrict__ b_dec, float* __restrict__ x2)
{
    const int b = blockIdx.x;
    const int a0 = threadIdx.x * 4;
    float4 acc = make_float4(0.f, 0.f, 0.f, 0.f);
    const float4* W4 = reinterpret_cast<const float4*>(W_dec);
    for (int e = 0; e < En; ++e) {
        float d = dinput[b * En + e];                 // uniform -> scalar load
        float4 w = W4[(e * An + a0) >> 2];
        acc.x = fmaf(d, w.x, acc.x);
        acc.y = fmaf(d, w.y, acc.y);
        acc.z = fmaf(d, w.z, acc.z);
        acc.w = fmaf(d, w.w, acc.w);
    }
    const float4 bd = reinterpret_cast<const float4*>(b_dec)[a0 >> 2];
    acc.x += bd.x; acc.y += bd.y; acc.z += bd.z; acc.w += bd.w;
    reinterpret_cast<float4*>(x2)[(b * An + a0) >> 2] = acc;
}

// ---------------------------------------------------------------------------
// K2: fused  scores[b,s] = relu(einput[b,s,:]@W_enc + b_enc + x2[b,:]) . W_att
//     + b_att.   bf16 WMMA GEMM, f32 accumulate, h never materialized.
//
// Workgroup = (b, 128-row s block), 256 threads = 8 waves; wave w owns rows
// [s0+16w, s0+16w+15]. LDS = exactly 320 KB (1 WG/WGP, 8 waves):
//   A block       : 128 rows x 512 bf16-pairs = 256 KB (staged once)
//   B tile x2 buf : fragment-ready bf16 pairs =  64 KB (double-buffered)
// Double buffering: global loads for tile nt+1 are issued BEFORE the K-loop
// of tile nt (results parked in VGPRs), packed/stored to the other LDS buffer
// after the epilogue -> W_enc L2 latency overlaps the 32 WMMAs; one barrier
// per iteration.
// ---------------------------------------------------------------------------
__global__ __launch_bounds__(256) void fused_scores_kernel(
    const float* __restrict__ einput, const float* __restrict__ W_enc,
    const float* __restrict__ b_enc,  const float* __restrict__ W_att,
    const float* __restrict__ b_att,  const float* __restrict__ x2,
    float* __restrict__ scores)
{
    extern __shared__ unsigned smem[];
    unsigned* ldsA  = smem;                     // 65536 u32
    unsigned* ldsB0 = smem + 65536;             //  8192 u32
    unsigned* ldsB1 = smem + 65536 + 8192;      //  8192 u32

    const int t    = threadIdx.x;
    const int lane = t & 31;
    const int wave = t >> 5;               // 0..7 : M-tile
    const int g    = lane >> 4;            // half-wave group
    const int n16  = lane & 15;            // column within 16
    const int b    = blockIdx.y;
    const int s0   = blockIdx.x * 128;

    // Per-thread B-staging coordinates (same for every tile): this thread
    // handles pairs p = t + j*256, j=0..31; p -> (column n, k-pair kp).
    // LDS target index for pair kp of column n:
    //   kb = kp>>4, gg = (kp>>3)&1, i = kp&7  ->  (((kb*2+gg)*16+n)<<3)+i

    // ---- Stage A: einput[b, s0..s0+127, 0..1023] -> bf16 pairs, row-major --
    const float2* ein2 = reinterpret_cast<const float2*>(einput);
    const long ein_base = ((long)b * Sn + s0) * (En / 2);
    for (int j = 0; j < 256; ++j) {
        int p  = t + j * 256;              // 0..65535
        int r  = p >> 9;                   // row 0..127
        int k2 = p & 511;                  // k-pair 0..511
        float2 v = ein2[ein_base + (long)r * 512 + k2];
        ldsA[r * 512 + k2] = pack_bf16(v.x, v.y);
    }

    // ---- Stage B tile 0 into ldsB0 (synchronous, once) ----
    #pragma unroll
    for (int j = 0; j < 32; ++j) {
        int p  = t + j * 256;
        int n  = p & 15;
        int kp = p >> 4;
        int k  = kp * 2;
        float f0 = W_enc[(long)k * An + n];
        float f1 = W_enc[(long)(k + 1) * An + n];
        int kb = kp >> 4, gg = (kp >> 3) & 1, i = kp & 7;
        ldsB0[(((kb * 2 + gg) * 16 + n) << 3) + i] = pack_bf16(f0, f1);
    }
    __syncthreads();   // A block + B tile 0 visible to all waves

    float part[8];
    #pragma unroll
    for (int j = 0; j < 8; ++j) part[j] = 0.f;
    const float batt = b_att[0];

    for (int nt = 0; nt < 64; ++nt) {
        const int n0 = nt * 16;
        unsigned* bcur = (nt & 1) ? ldsB1 : ldsB0;
        unsigned* bnxt = (nt & 1) ? ldsB0 : ldsB1;

        // ---- Issue global loads for tile nt+1 (latency hidden by K-loop) ---
        float pf[64];
        if (nt + 1 < 64) {
            const int n0n = n0 + 16;
            #pragma unroll
            for (int j = 0; j < 32; ++j) {
                int p  = t + j * 256;
                int n  = p & 15;
                int k  = (p >> 4) * 2;
                pf[2 * j]     = W_enc[(long)k * An + n0n + n];
                pf[2 * j + 1] = W_enc[(long)(k + 1) * An + n0n + n];
            }
        }

        // ---- K loop: 32 x v_wmma_f32_16x16x32_bf16 from bcur ----
        v8f c = {};
        const int rowA = wave * 16 + n16;
        #pragma unroll 4
        for (int kb = 0; kb < 32; ++kb) {
            union { v16bf v; uint4 q[2]; } afrag, bfrag;
            // A: lane holds row rowA, K = kb*32 + 16g + {0..15}
            const uint4* pa =
                reinterpret_cast<const uint4*>(ldsA + rowA * 512 + kb * 16 + 8 * g);
            afrag.q[0] = pa[0];
            afrag.q[1] = pa[1];
            // B: lane holds col n16, K = kb*32 + 16g + {0..15}
            const uint4* pb =
                reinterpret_cast<const uint4*>(bcur + (((kb * 2 + g) * 16 + n16) << 3));
            bfrag.q[0] = pb[0];
            bfrag.q[1] = pb[1];
            c = __builtin_amdgcn_wmma_f32_16x16x32_bf16(
                    false, afrag.v, false, bfrag.v, (short)0, c, false, false);
        }

        // ---- Fused epilogue: +b_enc +x2, ReLU, * W_att[col], accumulate ----
        const int col  = n0 + n16;
        const float bias = b_enc[col] + x2[b * An + col];
        const float wat  = W_att[col];
        #pragma unroll
        for (int j = 0; j < 8; ++j) {
            float v = c[j] + bias;
            v = fmaxf(v, 0.f);
            part[j] = fmaf(v, wat, part[j]);
        }

        // ---- Pack prefetched tile nt+1 into the other buffer ----
        if (nt + 1 < 64) {
            #pragma unroll
            for (int j = 0; j < 32; ++j) {
                int p  = t + j * 256;
                int n  = p & 15;
                int kp = p >> 4;
                int kb = kp >> 4, gg = (kp >> 3) & 1, i = kp & 7;
                bnxt[(((kb * 2 + gg) * 16 + n) << 3) + i] =
                    pack_bf16(pf[2 * j], pf[2 * j + 1]);
            }
        }
        __syncthreads();   // bnxt filled; bcur free for overwrite in nt+2
    }

    // Reduce over the 16 columns held by each 16-lane half (xor stays in-group)
    #pragma unroll
    for (int j = 0; j < 8; ++j) {
        float v = part[j];
        v += __shfl_xor(v, 1, 32);
        v += __shfl_xor(v, 2, 32);
        v += __shfl_xor(v, 4, 32);
        v += __shfl_xor(v, 8, 32);
        if (n16 == 0) {
            int row = j + 8 * g;           // C layout: vgpr j -> M = j + 8*(lane>=16)
            scores[(long)b * Sn + s0 + wave * 16 + row] = v + batt;
        }
    }
}

// ---------------------------------------------------------------------------
// K3: attn[b,:] = softmax(scores[b,:]) over S=2048. One block per b.
// ---------------------------------------------------------------------------
__global__ __launch_bounds__(256) void softmax_kernel(
    const float* __restrict__ scores, float* __restrict__ attn)
{
    __shared__ float red[256];
    const int b = blockIdx.x, t = threadIdx.x;
    float e[8];
    float mx = -3.402823466e38f;
    #pragma unroll
    for (int i = 0; i < 8; ++i) {
        e[i] = scores[b * Sn + t + i * 256];
        mx = fmaxf(mx, e[i]);
    }
    red[t] = mx; __syncthreads();
    for (int off = 128; off > 0; off >>= 1) {
        if (t < off) red[t] = fmaxf(red[t], red[t + off]);
        __syncthreads();
    }
    mx = red[0]; __syncthreads();
    float sum = 0.f;
    #pragma unroll
    for (int i = 0; i < 8; ++i) { e[i] = __expf(e[i] - mx); sum += e[i]; }
    red[t] = sum; __syncthreads();
    for (int off = 128; off > 0; off >>= 1) {
        if (t < off) red[t] += red[t + off];
        __syncthreads();
    }
    const float inv = 1.0f / red[0];
    #pragma unroll
    for (int i = 0; i < 8; ++i) attn[b * Sn + t + i * 256] = e[i] * inv;
}

// ---------------------------------------------------------------------------
// K4a: split-S partial weighted sums: partial[b,sc,e] = sum_{s in chunk} ein*attn
// ---------------------------------------------------------------------------
__global__ __launch_bounds__(256) void wsum_partial_kernel(
    const float* __restrict__ einput, const float* __restrict__ attn,
    float* __restrict__ partial)
{
    const int b = blockIdx.y, sc = blockIdx.x, t = threadIdx.x;
    const int e0 = t * 4;
    float4 acc = make_float4(0.f, 0.f, 0.f, 0.f);
    const float4* ein4 = reinterpret_cast<const float4*>(einput);
    const long base = (((long)b * Sn + sc * 128) * En + e0) >> 2;
    for (int s = 0; s < 128; ++s) {
        float a  = attn[b * Sn + sc * 128 + s];
        float4 v = ein4[base + (long)s * (En >> 2)];
        acc.x = fmaf(a, v.x, acc.x);
        acc.y = fmaf(a, v.y, acc.y);
        acc.z = fmaf(a, v.z, acc.z);
        acc.w = fmaf(a, v.w, acc.w);
    }
    reinterpret_cast<float4*>(partial)[(((b * 16 + sc) * En) + e0) >> 2] = acc;
}

// ---------------------------------------------------------------------------
// K4b: weights[b,e] = sum over 16 partials
// ---------------------------------------------------------------------------
__global__ __launch_bounds__(256) void wsum_reduce_kernel(
    const float* __restrict__ partial, float* __restrict__ weights)
{
    const int b = blockIdx.x, t = threadIdx.x;
    const int e0 = t * 4;
    float4 acc = make_float4(0.f, 0.f, 0.f, 0.f);
    const float4* p4 = reinterpret_cast<const float4*>(partial);
    #pragma unroll
    for (int sc = 0; sc < 16; ++sc) {
        float4 v = p4[(((b * 16 + sc) * En) + e0) >> 2];
        acc.x += v.x; acc.y += v.y; acc.z += v.z; acc.w += v.w;
    }
    reinterpret_cast<float4*>(weights)[(b * En + e0) >> 2] = acc;
}

// ---------------------------------------------------------------------------
extern "C" void kernel_launch(void* const* d_in, const int* in_sizes, int n_in,
                              void* d_out, int out_size, void* d_ws, size_t ws_size,
                              hipStream_t stream) {
    const float* einput = (const float*)d_in[0];
    const float* dinput = (const float*)d_in[1];
    const float* W_enc  = (const float*)d_in[2];
    const float* b_enc  = (const float*)d_in[3];
    const float* W_dec  = (const float*)d_in[4];
    const float* b_dec  = (const float*)d_in[5];
    const float* W_att  = (const float*)d_in[6];
    const float* b_att  = (const float*)d_in[7];

    float* weights_out = (float*)d_out;                  // [32*1024]
    float* attn_out    = (float*)d_out + Bn * En;        // [32*2048]

    float* ws      = (float*)d_ws;
    float* x2      = ws;                                 // 32768 floats
    float* scores  = ws + Bn * An;                       // 65536 floats
    float* partial = ws + Bn * An + Bn * Sn;             // 524288 floats

    dec_proj_kernel<<<dim3(Bn), dim3(256), 0, stream>>>(dinput, W_dec, b_dec, x2);

    // Exactly 320 KB dynamic LDS: 256 KB A-block + 2 x 32 KB B double buffer
    const size_t lds_bytes = (65536u + 2u * 8192u) * sizeof(unsigned);
    fused_scores_kernel<<<dim3(Sn / 128, Bn), dim3(256), lds_bytes, stream>>>(
        einput, W_enc, b_enc, W_att, b_att, x2, scores);

    softmax_kernel<<<dim3(Bn), dim3(256), 0, stream>>>(scores, attn_out);

    wsum_partial_kernel<<<dim3(16, Bn), dim3(256), 0, stream>>>(einput, attn_out,
                                                                partial);
    wsum_reduce_kernel<<<dim3(Bn), dim3(256), 0, stream>>>(partial, weights_out);
}
